// DeepPotE3Embedding_21423296873077
// MI455X (gfx1250) — matrix operand order repaired and checked by reading, not compiled, not run on hardware
//
#include <hip/hip_runtime.h>

#define N_ATOMS   50000
#define N_EDGES   1600000
#define N_ANGLES  1500000
#define N_SPECIES 16
#define DIM       64

typedef __attribute__((ext_vector_type(16))) _Float16 v16h;
typedef __attribute__((ext_vector_type(8)))  _Float16 v8h;
typedef __attribute__((ext_vector_type(8)))  float    v8f;

// tanh via hardware transcendental when available
#if __has_builtin(__builtin_amdgcn_tanhf)
__device__ __forceinline__ float tanh_hw(float x) { return __builtin_amdgcn_tanhf(x); }
#else
__device__ __forceinline__ float tanh_hw(float x) {
    return 2.0f * __builtin_amdgcn_rcpf(1.0f + __expf(-2.0f * x)) - 1.0f;
}
#endif
// silu(x) = x*sigmoid(x) = h + h*tanh(h), h = x/2  (1 TRANS op per element)

// ---------------------------------------------------------------------------
// Kernel 1: per-edge preprocessing.
//   Rij[e] = (sw/d, sw*vx/d^2, sw*vy/d^2, sw*vz/d^2)   (float4)
//   es[e]  = species[edge_dst[e]]                      (int)
// ---------------------------------------------------------------------------
__global__ __launch_bounds__(256) void prep_edges(
    const int*   __restrict__ species,
    const int*   __restrict__ edge_dst,
    const float* __restrict__ dist,
    const float* __restrict__ sw,
    const float* __restrict__ vec,
    float4*      __restrict__ Rij,
    int*         __restrict__ es)
{
    int e = blockIdx.x * blockDim.x + threadIdx.x;
    if (e >= N_EDGES) return;
    float d  = dist[e];
    float s  = sw[e];
    float vx = vec[3 * e + 0];
    float vy = vec[3 * e + 1];
    float vz = vec[3 * e + 2];
    float inv = 1.0f / d;
    float sij = s * inv;
    float f   = sij * inv;
    Rij[e] = make_float4(sij, f * vx, f * vy, f * vz);
    es[e]  = species[edge_dst[e]];
}

// ---------------------------------------------------------------------------
// Kernel 2: species-pair table for layer 0, pre-scaled by 1/2 for the
// tanh-silu form: T[(s1*16+s2)*64+j] = 0.5*(W0[1+s1][j] + W0[17+s2][j] + b0[j])
// ---------------------------------------------------------------------------
__global__ __launch_bounds__(256) void prep_pair_table(
    const float* __restrict__ W0,
    const float* __restrict__ b0,
    float*       __restrict__ T)
{
    int idx = blockIdx.x * blockDim.x + threadIdx.x;   // 0 .. 16383
    if (idx >= 256 * 64) return;
    int pair = idx >> 6;
    int j    = idx & 63;
    int s1   = pair >> 4;
    int s2   = pair & 15;
    T[idx] = 0.5f * (W0[(1 + s1) * 64 + j] + W0[(17 + s2) * 64 + j] + b0[j]);
}

// ---------------------------------------------------------------------------
// Kernel 3: angle MLP + segment-sum via WMMA.
// One wave32 handles 8 angles = 16 MLP rows (rows 0-7: (src,dst) ordering,
// rows 8-15: (dst,src)). Dense 64x64 layers: v_wmma_f32_16x16x32_f16,
// 8 back-to-back WMMAs per layer into 4 accumulators, weights in registers.
// ---------------------------------------------------------------------------
__global__ __launch_bounds__(256) void angle_mlp(
    const float4* __restrict__ Rij,
    const int*    __restrict__ es,
    const float*  __restrict__ T,       // 256x64 half-scaled pair table
    const int*    __restrict__ angle_src,
    const int*    __restrict__ angle_dst,
    const int*    __restrict__ central,
    const float*  __restrict__ W0,      // only row 0 needed here
    const float*  __restrict__ W1, const float* __restrict__ b1,
    const float*  __restrict__ W2, const float* __restrict__ b2,
    const float*  __restrict__ W3, const float* __restrict__ b3,
    float*        __restrict__ out,
    int nTiles)
{
    // ---- static LDS (~43 KB / block) ----
    __shared__ float                  sw0[64];          // 0.5 * W0 row 0
    __shared__ float                  sB[3][64];        // 0.5*b1, 0.5*b2, b3
    __shared__ __align__(16) _Float16 sWt[3][64 * 64];  // W1..W3 transposed, f16
    __shared__ __align__(16) _Float16 sH[8][16 * 64];   // per-wave activations
    __shared__ float                  sTh[8][8];        // per-wave theta
    __shared__ int                    sCe[8][8];        // per-wave central atom
    __shared__ int                    sP1[8][8];        // pair (s1,s2)
    __shared__ int                    sP2[8][8];        // pair (s2,s1)

    const int tid = threadIdx.x;

    // ---- stage weights into LDS (once per block) ----
    for (int i = tid; i < 64; i += 256) {
        sw0[i]   = 0.5f * W0[i];
        sB[0][i] = 0.5f * b1[i];
        sB[1][i] = 0.5f * b2[i];
        sB[2][i] = b3[i];
    }
    {
        const float* Ws[3] = { W1, W2, W3 };
        for (int L = 0; L < 3; ++L)
            for (int i = tid; i < 4096; i += 256) {
                int n = i >> 6, k = i & 63;
                sWt[L][n * 64 + k] = (_Float16)Ws[L][k * 64 + n]; // transpose
            }
    }
    __syncthreads();

    const int wv   = tid >> 5;      // wave in block (0..7)
    const int lane = tid & 31;
    const int nLow = lane & 15;     // N index within a 16-tile / row M for A
    const int hi   = lane >> 4;     // lane half (0/1)

    // ---- hoist all B-matrix WMMA fragments into registers (24 x v16h) ----
    v16h wB[3][4][2];
#pragma unroll
    for (int L = 0; L < 3; ++L)
#pragma unroll
        for (int nt = 0; nt < 4; ++nt)
#pragma unroll
            for (int kb = 0; kb < 2; ++kb) {
                int n    = nt * 16 + nLow;
                int koff = kb * 32 + hi * 16;
                const v8h* p = (const v8h*)&sWt[L][n * 64 + koff];
                union { v16h v; v8h h[2]; } u;
                u.h[0] = p[0];
                u.h[1] = p[1];
                wB[L][nt][kb] = u.v;
            }

    const int gw = blockIdx.x * 8 + wv;   // global wave id
    const int nw = gridDim.x * 8;

    for (int tile = gw; tile < nTiles; tile += nw) {
        const int a0 = tile * 8;

        // ---- gather phase: lanes 0..7 each own one angle ----
        if (lane < 8) {
            int  a  = a0 + lane;
            bool ok = (a < N_ANGLES);
            int  aa = ok ? a : 0;
            int  e1 = angle_src[aa];
            int  e2 = angle_dst[aa];
            float4 r1 = Rij[e1];
            float4 r2 = Rij[e2];
            float th = r1.x * r2.x + r1.y * r2.y + r1.z * r2.z + r1.w * r2.w;
            int s1 = es[e1];
            int s2 = es[e2];
            sTh[wv][lane] = ok ? th : 0.0f;
            sCe[wv][lane] = ok ? central[aa] : 0;
            sP1[wv][lane] = s1 * 16 + s2;
            sP2[wv][lane] = s2 * 16 + s1;
        }
        asm volatile("s_wait_dscnt 0" ::: "memory");

        // ---- layer 0 (sparse): h = theta*(W0row0/2) + T[pair]  (already /2)
        // lane covers row m = lane%16, columns [hi*32, hi*32+32)
        {
            const int m  = nLow;
            const int j0 = hi * 32;
            const int ai = m & 7;
            float th   = sTh[wv][ai];
            int   pair = (m < 8) ? sP1[wv][ai] : sP2[wv][ai];
            const float* trow = &T[pair * 64];
#pragma unroll
            for (int jj = 0; jj < 8; ++jj) {
                int j = j0 + jj * 4;
                float4 w0 = *(const float4*)&sw0[j];
                float4 tv = *(const float4*)&trow[j];
                float  h[4], t[4];
                h[0] = th * w0.x + tv.x;
                h[1] = th * w0.y + tv.y;
                h[2] = th * w0.z + tv.z;
                h[3] = th * w0.w + tv.w;
#pragma unroll
                for (int q = 0; q < 4; ++q) t[q] = tanh_hw(h[q]);
                union { float2 f2; _Float16 hh[4]; } u;
#pragma unroll
                for (int q = 0; q < 4; ++q)
                    u.hh[q] = (_Float16)(h[q] * t[q] + h[q]);
                *(float2*)&sH[wv][m * 64 + j] = u.f2;
            }
        }
        asm volatile("s_wait_dscnt 0" ::: "memory");

        // ---- dense layers 1..3 via WMMA ----
#pragma unroll
        for (int L = 0; L < 3; ++L) {
            // A fragments (16x32 f16): lane: M = l%16,
            // elems 0..15 = h[m][kb*32 + (l/16)*16 ..] (contiguous)
            v16h aA[2];
#pragma unroll
            for (int kb = 0; kb < 2; ++kb) {
                int kh = kb * 32 + hi * 16;
                union { v16h v; v8h h[2]; } u;
                u.h[0] = *(const v8h*)&sH[wv][nLow * 64 + kh];
                u.h[1] = *(const v8h*)&sH[wv][nLow * 64 + kh + 8];
                aA[kb] = u.v;
            }
            // preload per-n biases (overlaps with WMMA issue)
            float bias[4];
#pragma unroll
            for (int nt = 0; nt < 4; ++nt) bias[nt] = sB[L][nt * 16 + nLow];
            asm volatile("" ::: "memory");

            // 8 back-to-back WMMAs into 4 accumulators (one hazard window)
            v8f c[4];
#pragma unroll
            for (int nt = 0; nt < 4; ++nt)
                c[nt] = __builtin_amdgcn_wmma_f32_16x16x32_f16(
                    false, aA[0], false, wB[L][nt][0], (short)0, v8f{}, false, false);
#pragma unroll
            for (int nt = 0; nt < 4; ++nt)
                c[nt] = __builtin_amdgcn_wmma_f32_16x16x32_f16(
                    false, aA[1], false, wB[L][nt][1], (short)0, c[nt], false, false);

            if (L < 2) {
#pragma unroll
                for (int nt = 0; nt < 4; ++nt) {
                    int n = nt * 16 + nLow;
                    // h = c/2 + b/2 ; silu = h + h*tanh(h)
                    float h[8], t[8];
#pragma unroll
                    for (int v = 0; v < 8; ++v) h[v] = c[nt][v] * 0.5f + bias[nt];
#pragma unroll
                    for (int v = 0; v < 8; ++v) t[v] = tanh_hw(h[v]);
#pragma unroll
                    for (int v = 0; v < 8; ++v) {
                        int mm = v + hi * 8;            // C layout: M = v + 8*(l/16)
                        sH[wv][mm * 64 + n] = (_Float16)(h[v] * t[v] + h[v]);
                    }
                }
                asm volatile("s_wait_dscnt 0" ::: "memory");
            } else {
                // final layer: fold the two orderings and scatter-add
#pragma unroll
                for (int nt = 0; nt < 4; ++nt) {
                    int n = nt * 16 + nLow;
#pragma unroll
                    for (int v = 0; v < 8; ++v) {
                        float g = c[nt][v] + bias[nt];  // rows: v (A) / v+8 (B)
                        g += __shfl_xor(g, 16, 32);     // G_A + G_B at (angle v, col n)
                        if (hi == 0) {
                            float thv = sTh[wv][v];
                            int   ca  = sCe[wv][v];
                            __hip_atomic_fetch_add(&out[ca * 64 + n], g * thv,
                                                   __ATOMIC_RELAXED,
                                                   __HIP_MEMORY_SCOPE_AGENT);
                        }
                    }
                }
            }
        }
    }
}

extern "C" void kernel_launch(void* const* d_in, const int* in_sizes, int n_in,
                              void* d_out, int out_size, void* d_ws, size_t ws_size,
                              hipStream_t stream) {
    const int*   species   = (const int*)d_in[0];
    const int*   edge_dst  = (const int*)d_in[1];
    const float* distances = (const float*)d_in[2];
    const float* switch_   = (const float*)d_in[3];
    const float* vec       = (const float*)d_in[4];
    const int*   angle_src = (const int*)d_in[5];
    const int*   angle_dst = (const int*)d_in[6];
    const int*   central   = (const int*)d_in[7];
    const float* W0 = (const float*)d_in[8];
    const float* b0 = (const float*)d_in[9];
    const float* W1 = (const float*)d_in[10];
    const float* b1 = (const float*)d_in[11];
    const float* W2 = (const float*)d_in[12];
    const float* b2 = (const float*)d_in[13];
    const float* W3 = (const float*)d_in[14];
    const float* b3 = (const float*)d_in[15];
    float* out = (float*)d_out;

    // workspace layout: float4 Rij[N_EDGES] | int es[N_EDGES] | float T[256*64]
    float4* Rij = (float4*)d_ws;
    int*    es  = (int*)((char*)d_ws + (size_t)N_EDGES * sizeof(float4));
    float*  T   = (float*)((char*)d_ws + (size_t)N_EDGES * (sizeof(float4) + sizeof(int)));

    // zero the segment-sum output (harness poisons it)
    hipMemsetAsync(d_out, 0, (size_t)out_size * sizeof(float), stream);

    prep_edges<<<(N_EDGES + 255) / 256, 256, 0, stream>>>(
        species, edge_dst, distances, switch_, vec, Rij, es);
    prep_pair_table<<<(256 * 64 + 255) / 256, 256, 0, stream>>>(W0, b0, T);

    const int nTiles = (N_ANGLES + 7) / 8;       // 187500
    int nBlocks = (nTiles + 7) / 8;
    if (nBlocks > 1536) nBlocks = 1536;          // persistent waves, amortize staging
    angle_mlp<<<nBlocks, 256, 0, stream>>>(
        Rij, es, T, angle_src, angle_dst, central,
        W0, W1, b1, W2, b2, W3, b3, out, nTiles);
}
